// FCOSModule_66408784331532
// MI455X (gfx1250) — matrix operand order repaired and checked by reading, not compiled, not run on hardware
//
#include <hip/hip_runtime.h>

// ---------------------------------------------------------------------------
// FCOS head on MI455X (gfx1250): implicit-GEMM conv3x3 via v_wmma_f32_16x16x32_bf16
// Double-buffered LDS staging so im2col gather overlaps the WMMA chain.
// ---------------------------------------------------------------------------

typedef __attribute__((ext_vector_type(16))) __bf16 v16bf;
typedef __attribute__((ext_vector_type(8)))  float  v8f;

#define IC      256
#define KTOT    2304          // 256 * 9
#define NKCH    72            // 2304 / 32
#define CG      8             // K-chunks staged per group
#define NGRP    9             // 72 / 8
#define TOTLOC  17064         // total locations per image

__device__ __forceinline__ unsigned short f2bf(float f) {
    unsigned int u = __float_as_uint(f);
    u += 0x7FFFu + ((u >> 16) & 1u);      // round-to-nearest-even
    return (unsigned short)(u >> 16);
}

// ---------------------------------------------------------------------------
// Weight repack: fp32 [O][256][3][3] -> bf16 in WMMA A-fragment order.
// A 16x32 bf16 fragment: lane L holds row M = L&15; element idx holds
// K = idx + (idx>=8 ? 8 : 0) + 8*(L>>4).   Flat k = i*9 + ky*3 + kx.
// mode 0: single weight tensor with Ovalid rows.
// mode 1: combined box(4 rows from W1) + ctr(1 row from W2), rest zero.
// ---------------------------------------------------------------------------
__global__ void repack_weights(const float* __restrict__ W1,
                               const float* __restrict__ W2,
                               unsigned short* __restrict__ Out,
                               int Ovalid, int nmt, int mode)
{
    int d = blockIdx.x * 256 + threadIdx.x;
    int total = nmt * NKCH * 512;
    if (d >= total) return;
    int idx  = d & 15;
    int L    = (d >> 4) & 31;
    int c    = (d >> 9) % NKCH;
    int mt   = d / (NKCH * 512);
    int hf   = L >> 4;
    int m    = mt * 16 + (L & 15);
    int k    = c * 32 + idx + (idx >= 8 ? 8 : 0) + hf * 8;
    float v = 0.f;
    if (mode == 0) {
        if (m < Ovalid) v = W1[m * KTOT + k];
    } else {
        if (m < 4)       v = W1[m * KTOT + k];
        else if (m == 4) v = W2[k];
    }
    Out[d] = f2bf(v);
}

__global__ void make_head_bias(const float* __restrict__ box_b,
                               const float* __restrict__ ctr_b,
                               float* __restrict__ out)
{
    int i = threadIdx.x;
    if (i < 4)       out[i] = box_b[i];
    else if (i == 4) out[i] = ctr_b[0];
    else if (i < 16) out[i] = 0.f;
}

// ---------------------------------------------------------------------------
// Implicit-GEMM conv3x3 (SAME), WMMA bf16.
// Block = 512 threads (16 waves). Each block: one (b, y, x0..x0+15) tile.
// Wave w computes output channels [16w, 16w+16).  K staged into a
// double-buffered LDS tile in groups of 8 chunks, laid out in B-fragment
// order (lane = half*16 + n, element idx holds K = idx + (idx>=8 ? 8 : 0) + 8*half).
// MODE 0: tower conv  -> Y[b,o,y,x] = acc + bias[o]
// MODE 1: cls head    -> out[b, o,  loc]
// MODE 2: box/ctr head-> out[b,16+o,loc] = relu((acc+b)*scale)*stride (o<4)
//                        out[b,20,  loc] = acc+b                      (o==4)
// ---------------------------------------------------------------------------
template<int MODE>
__global__ __launch_bounds__(512, 1)
void conv3x3_wmma(const float* __restrict__ X,
                  const unsigned short* __restrict__ Wf,
                  const float* __restrict__ bias,
                  float* __restrict__ Out,
                  int H, int W, int ntx, int nmt,
                  const float* __restrict__ scales, int level,
                  int loff, float stride_f)
{
    __shared__ __align__(32) unsigned short Bs[2][CG * 512];   // 2 x 8 KB

    int blk = blockIdx.x;
    int tx  = blk % ntx;
    int t2  = blk / ntx;
    int y0  = t2 % H;
    int b   = t2 / H;
    int x0  = tx * 16;

    int tid  = threadIdx.x;
    int lane = tid & 31;
    int wave = tid >> 5;
    int hf   = (lane >> 4) & 1;
    int nn   = lane & 15;

    // im2col gather of 8 K-chunks (group cg) into LDS buffer bufsel,
    // values written directly in B-fragment layout.
    auto stage = [&](int cg, int bufsel) {
        #pragma unroll
        for (int e = 0; e < 8; ++e) {
            int flat = tid + (e << 9);        // 0..4095
            int n  = flat & 15;               // fastest -> coalesced x reads
            int kk = (flat >> 4) & 31;
            int cl = flat >> 9;
            int k  = ((cg << 3) + cl) * 32 + kk;
            int i  = k / 9;
            int rr = k - i * 9;
            int yy = y0 + rr / 3 - 1;
            int xx = x0 + n + (rr % 3) - 1;
            float v = 0.f;
            if (xx >= 0 && xx < W && yy >= 0 && yy < H)
                v = X[((b * IC + i) * H + yy) * W + xx];
            int h2  = (kk >> 3) & 1;
            int idx = (kk & 7) + ((kk >> 4) << 3);
            Bs[bufsel][(((cl << 5) + (h2 << 4) + n) << 4) + idx] = f2bf(v);
        }
    };

    v8f acc = {};

    stage(0, 0);
    __syncthreads();

    for (int g = 0; g < NGRP; ++g) {
        int cur = g & 1;
        // issue next group's gather before computing on the current one
        if (g + 1 < NGRP) stage(g + 1, cur ^ 1);

        if (wave < nmt) {
            #pragma unroll
            for (int cl = 0; cl < CG; ++cl) {
                int c = (g << 3) + cl;
                v16bf a = *(const v16bf*)(Wf +
                            ((((size_t)wave * NKCH + c) * 32 + lane) << 4));
                v16bf bm = *(const v16bf*)(&Bs[cur][((cl << 5) + lane) << 4]);
                acc = __builtin_amdgcn_wmma_f32_16x16x32_bf16(
                          false, a, false, bm, (short)0, acc, false, false);
            }
        }
        __syncthreads();
    }

    // ---- epilogue ----
    if (wave >= nmt) return;
    int x = x0 + nn;
    if (x >= W) return;

    #pragma unroll
    for (int r = 0; r < 8; ++r) {
        int o = (MODE == 0 ? wave * 16 : 0) + r + 8 * hf;
        float v = acc[r] + bias[o];
        if (MODE == 0) {
            Out[(((size_t)b * IC + o) * H + y0) * W + x] = v;
        } else if (MODE == 1) {
            Out[((size_t)(b * 21 + o)) * TOTLOC + loff + y0 * W + x] = v;
        } else {
            if (o < 4) {
                float s = scales[level];
                float t = v * s;
                t = t > 0.f ? t : 0.f;
                Out[((size_t)(b * 21 + 16 + o)) * TOTLOC + loff + y0 * W + x]
                    = t * stride_f;
            } else if (o == 4) {
                Out[((size_t)(b * 21 + 20)) * TOTLOC + loff + y0 * W + x] = v;
            }
        }
    }
}

// ---------------------------------------------------------------------------
// GroupNorm: stats (one block per (b,group); group = 8 contiguous channels)
// ---------------------------------------------------------------------------
__global__ __launch_bounds__(256, 1)
void gn_stats(const float* __restrict__ Y, float* __restrict__ stats, int HW)
{
    int bg = blockIdx.x;                 // b*32 + g
    size_t base = (size_t)bg * 8 * HW;   // (b*256 + g*8)*HW
    int cnt = 8 * HW;
    float s1 = 0.f, s2 = 0.f;
    for (int i = threadIdx.x; i < cnt; i += 256) {
        float v = Y[base + i];
        s1 += v; s2 += v * v;
    }
    __shared__ float sh1[256], sh2[256];
    sh1[threadIdx.x] = s1; sh2[threadIdx.x] = s2;
    __syncthreads();
    for (int s = 128; s > 0; s >>= 1) {
        if (threadIdx.x < s) {
            sh1[threadIdx.x] += sh1[threadIdx.x + s];
            sh2[threadIdx.x] += sh2[threadIdx.x + s];
        }
        __syncthreads();
    }
    if (threadIdx.x == 0) {
        float m   = sh1[0] / (float)cnt;
        float var = sh2[0] / (float)cnt - m * m;
        stats[bg * 2 + 0] = m;
        stats[bg * 2 + 1] = rsqrtf(var + 1e-5f);
    }
}

__global__ void gn_apply(const float* __restrict__ Y, float* __restrict__ Xo,
                         const float* __restrict__ gamma,
                         const float* __restrict__ beta,
                         const float* __restrict__ stats,
                         int HW, long N)
{
    long i = (long)blockIdx.x * 256 + threadIdx.x;
    if (i >= N) return;
    int c  = (int)(i / HW) & 255;
    int bg = (int)(i / ((long)HW * 8));
    float m  = stats[bg * 2 + 0];
    float rs = stats[bg * 2 + 1];
    float v = (Y[i] - m) * rs * gamma[c] + beta[c];
    Xo[i] = v > 0.f ? v : 0.f;
}

__global__ void locs_kernel(float* __restrict__ locs, int H, int W,
                            int stride, int loff)
{
    int i = blockIdx.x * 256 + threadIdx.x;
    if (i >= H * W) return;
    int y = i / W, x = i % W;
    locs[(size_t)(loff + i) * 2 + 0] = (float)(x * stride + stride / 2);
    locs[(size_t)(loff + i) * 2 + 1] = (float)(y * stride + stride / 2);
}

// ---------------------------------------------------------------------------
// Orchestration
// d_in: 0..4 = f0..f4; 5..20 = cls_tower (W,b,g,be)x4; 21..36 = bbox_tower;
// 37 cls_W, 38 cls_b, 39 box_W, 40 box_b, 41 ctr_W, 42 ctr_b, 43 scales.
// ---------------------------------------------------------------------------
extern "C" void kernel_launch(void* const* d_in, const int* in_sizes, int n_in,
                              void* d_out, int out_size, void* d_ws, size_t ws_size,
                              hipStream_t stream)
{
    (void)in_sizes; (void)n_in; (void)out_size; (void)ws_size;

    const int Hs[5]      = {100, 50, 25, 13, 7};
    const int Wss[5]     = {128, 64, 32, 16, 8};
    const int ntxs[5]    = {8, 4, 2, 1, 1};
    const int strides[5] = {8, 16, 32, 64, 128};
    const int loffs[5]   = {0, 12800, 16000, 16800, 17008};

    char* ws = (char*)d_ws;
    const size_t xsz = (size_t)8 * 256 * 100 * 128 * sizeof(float);
    float* Xbuf = (float*)ws;
    float* Ybuf = (float*)(ws + xsz);
    float* stats = (float*)(ws + 2 * xsz);
    float* headbias = (float*)(ws + 2 * xsz + 2048);
    unsigned short* wrep = (unsigned short*)(ws + 2 * xsz + 4096);
    const size_t TW  = (size_t)16 * NKCH * 512;   // elements per tower layer
    const size_t HWE = (size_t)NKCH * 512;        // elements per head
    unsigned short* wr_t[8];
    for (int i = 0; i < 8; ++i) wr_t[i] = wrep + (size_t)i * TW;
    unsigned short* wr_ch = wrep + 8 * TW;
    unsigned short* wr_bh = wr_ch + HWE;

    // ---- repack weights (bf16, A-fragment layout) ----
    int gT = (int)((TW + 255) / 256);
    int gH = (int)((HWE + 255) / 256);
    for (int j = 0; j < 4; ++j) {
        repack_weights<<<gT, 256, 0, stream>>>((const float*)d_in[5 + 4 * j],
                                               nullptr, wr_t[j], 256, 16, 0);
        repack_weights<<<gT, 256, 0, stream>>>((const float*)d_in[21 + 4 * j],
                                               nullptr, wr_t[4 + j], 256, 16, 0);
    }
    repack_weights<<<gH, 256, 0, stream>>>((const float*)d_in[37], nullptr,
                                           wr_ch, 16, 1, 0);
    repack_weights<<<gH, 256, 0, stream>>>((const float*)d_in[39],
                                           (const float*)d_in[41],
                                           wr_bh, 16, 1, 1);
    make_head_bias<<<1, 16, 0, stream>>>((const float*)d_in[40],
                                         (const float*)d_in[42], headbias);

    float* out  = (float*)d_out;
    float* locs = out + (size_t)8 * 21 * TOTLOC;
    const float* scales = (const float*)d_in[43];

    for (int l = 0; l < 5; ++l) {
        int H = Hs[l], W = Wss[l], ntx = ntxs[l];
        int HWp = H * W;
        long Ntot = (long)8 * 256 * HWp;
        int convGrid = 8 * H * ntx;
        int gnApplyGrid = (int)((Ntot + 255) / 256);

        for (int tower = 0; tower < 2; ++tower) {
            int pbase = (tower == 0) ? 5 : 21;
            const float* xin = (const float*)d_in[l];
            for (int j = 0; j < 4; ++j) {
                conv3x3_wmma<0><<<convGrid, 512, 0, stream>>>(
                    xin, wr_t[tower * 4 + j],
                    (const float*)d_in[pbase + 4 * j + 1], Ybuf,
                    H, W, ntx, 16, nullptr, 0, 0, 0.f);
                gn_stats<<<256, 256, 0, stream>>>(Ybuf, stats, HWp);
                gn_apply<<<gnApplyGrid, 256, 0, stream>>>(
                    Ybuf, Xbuf,
                    (const float*)d_in[pbase + 4 * j + 2],
                    (const float*)d_in[pbase + 4 * j + 3],
                    stats, HWp, Ntot);
                xin = Xbuf;
            }
            if (tower == 0) {
                conv3x3_wmma<1><<<convGrid, 512, 0, stream>>>(
                    Xbuf, wr_ch, (const float*)d_in[38], out,
                    H, W, ntx, 1, nullptr, 0, loffs[l], 0.f);
            } else {
                conv3x3_wmma<2><<<convGrid, 512, 0, stream>>>(
                    Xbuf, wr_bh, headbias, out,
                    H, W, ntx, 1, scales, l, loffs[l], (float)strides[l]);
            }
        }
        locs_kernel<<<(HWp + 255) / 256, 256, 0, stream>>>(
            locs, H, W, strides[l], loffs[l]);
    }
}
